// RCNN_2121713844584
// MI455X (gfx1250) — matrix operand order repaired and checked by reading, not compiled, not run on hardware
//
#include <hip/hip_runtime.h>

typedef __bf16 bf16;
typedef bf16 bf16x8  __attribute__((ext_vector_type(8)));
typedef bf16 bf16x16 __attribute__((ext_vector_type(16)));
typedef float v8f    __attribute__((ext_vector_type(8)));
typedef int  v4i     __attribute__((ext_vector_type(4)));

#define B_    64
#define L_    512
#define D_    300
#define Dp    320          // D padded to multiple of WMMA K=32
#define KC    960          // conv GEMM K: 3 taps x 320 (contiguous in padded layout)
#define F_    256
#define FF_   2048
#define NROI  2048
#define LPAD  514          // L + 2 zero rows for conv halo
#define BPITCH 40          // LDS B-tile row pitch in bf16 (32 + 8 pad, bank-friendly)

#define SHUF16(v0, v1) __builtin_shufflevector((v0), (v1), 0,1,2,3,4,5,6,7,8,9,10,11,12,13,14,15)
#define WMMA_BF16(a, b, c) __builtin_amdgcn_wmma_f32_16x16x32_bf16(false, (a), false, (b), (short)0, (c), false, false)

#if defined(__AMDGCN__) && __has_builtin(__builtin_amdgcn_global_load_async_to_lds_b128)
#define HAVE_ASYNC_LDS 1
#endif

// Stage 16 bytes global -> LDS. Async path: no VGPR round-trip, tracked on ASYNCcnt.
static __device__ __forceinline__ void stage_b128(bf16* ldsDst, const bf16* gsrc) {
#ifdef HAVE_ASYNC_LDS
  __builtin_amdgcn_global_load_async_to_lds_b128((v4i*)gsrc, (v4i*)ldsDst, 0, 0);
#else
  *(bf16x8*)ldsDst = *(const bf16x8*)gsrc;   // sync fallback: global load + ds_store
#endif
}

static __device__ __forceinline__ void stage_wait() {
#ifdef HAVE_ASYNC_LDS
#if __has_builtin(__builtin_amdgcn_s_wait_asynccnt)
  __builtin_amdgcn_s_wait_asynccnt(0);
#else
  asm volatile("s_wait_asynccnt 0x0" ::: "memory");
#endif
#endif
}

// B fragment (32x16 bf16) from LDS slab: lane half selects K 0-7,16-23 / 8-15,24-31.
static __device__ __forceinline__ bf16x16 lds_b_frag(const bf16* base, int q, int lane, int bOff) {
  const bf16* p = base + (q * 16 + (lane & 15)) * BPITCH + bOff;
  bf16x8 v0 = *(const bf16x8*)p;
  bf16x8 v1 = *(const bf16x8*)(p + 16);
  return SHUF16(v0, v1);
}

static __device__ __forceinline__ bf16x16 load_a16(const bf16* p) {
  bf16x8 v0 = *(const bf16x8*)p;
  bf16x8 v1 = *(const bf16x8*)(p + 16);
  return SHUF16(v0, v1);
}

// One K-step: 4 B fragments (pipelined) x 2 A fragments = 8 WMMAs.
static __device__ __forceinline__ void compute_step(const bf16* sbuf, bf16x16 a0, bf16x16 a1,
                                                    int lane, int bOff, v8f (&acc)[2][4]) {
  bf16x16 bq = lds_b_frag(sbuf, 0, lane, bOff);
#pragma unroll
  for (int q = 0; q < 4; ++q) {
    bf16x16 bn = bq;
    if (q < 3) bn = lds_b_frag(sbuf, q + 1, lane, bOff);
    acc[0][q] = WMMA_BF16(a0, bq, acc[0][q]);
    acc[1][q] = WMMA_BF16(a1, bq, acc[1][q]);
    bq = bn;
  }
}

// Generic pipelined GEMM core. A addressing and B staging are LINEAR in step s
// (stride 32 elements). NS must be even. acc = 32(M) x 64(N) per wave.
// NOTE: prefetches run one step past the K range (operands must be tail-padded).
static __device__ __forceinline__ void gemm_core(int NS,
                                                 const bf16* pA0, const bf16* pA1,
                                                 const bf16* pB,
                                                 bf16* buf0, bf16* buf1, int sOff,
                                                 int lane, int bOff, v8f (&acc)[2][4]) {
  stage_b128(buf0 + sOff, pB);
  bf16x16 aE0 = load_a16(pA0), aE1 = load_a16(pA1);
  stage_wait();
  __syncthreads();
  for (int s = 0; s < NS; s += 2) {
    // phase A: prefetch step s+1 (A regs + async B slab into buf1), compute step s
    bf16x16 aO0 = load_a16(pA0 + (s + 1) * 32);
    bf16x16 aO1 = load_a16(pA1 + (s + 1) * 32);
    stage_b128(buf1 + sOff, pB + (s + 1) * 32);
    compute_step(buf0, aE0, aE1, lane, bOff, acc);
    stage_wait();
    __syncthreads();
    // phase B: prefetch step s+2 into buf0 (unconditional; harmless tail overread)
    aE0 = load_a16(pA0 + (s + 2) * 32);
    aE1 = load_a16(pA1 + (s + 2) * 32);
    stage_b128(buf0 + sOff, pB + (s + 2) * 32);
    compute_step(buf1, aO0, aO1, lane, bOff, acc);
    stage_wait();
    __syncthreads();
  }
}

// ---------------- prep kernels ----------------
__global__ void k_prep_sentence(const float* __restrict__ s, bf16* __restrict__ out) {
  int idx = blockIdx.x * blockDim.x + threadIdx.x;
  const int total = B_ * LPAD * Dp;
  if (idx >= total) return;
  int d = idx % Dp;
  int r = idx / Dp;
  int lp = r % LPAD;
  int b  = r / LPAD;
  int l  = lp - 1;
  float v = 0.f;
  if (l >= 0 && l < L_ && d < D_) v = s[((size_t)b * L_ + l) * D_ + d];
  out[idx] = (bf16)v;
}

// conv_w [F,1,3,D] f32 -> [f][960] bf16 (taps concatenated, zero D-pad per tap)
__global__ void k_prep_convw(const float* __restrict__ w, bf16* __restrict__ out) {
  int idx = blockIdx.x * blockDim.x + threadIdx.x;
  const int total = F_ * KC;
  if (idx >= total) return;
  int rem = idx % KC;
  int f = idx / KC;
  int k = rem / Dp;
  int d = rem % Dp;
  float v = (d < D_) ? w[(size_t)f * (3 * D_) + k * D_ + d] : 0.f;
  out[idx] = (bf16)v;
}

__global__ void k_collapse_w(const float* __restrict__ w, bf16* __restrict__ outT) {
  int idx = blockIdx.x * blockDim.x + threadIdx.x;
  const int total = FF_ * F_;
  if (idx >= total) return;
  int f = idx % F_;
  int j = idx / F_;
  float s = 0.f;
#pragma unroll
  for (int p = 0; p < 8; ++p) s += w[(size_t)(f * 8 + p) * FF_ + j];
  outT[(size_t)j * F_ + f] = (bf16)s;
}

__global__ void k_transpose_pad(const float* __restrict__ src, bf16* __restrict__ dst,
                                int rowsPad, int nValid) {
  int idx = blockIdx.x * blockDim.x + threadIdx.x;
  int total = rowsPad * FF_;
  if (idx >= total) return;
  int k = idx % FF_;
  int c = idx / FF_;
  float v = (c < nValid) ? src[(size_t)k * nValid + c] : 0.f;
  dst[(size_t)c * FF_ + k] = (bf16)v;
}

// ---------------- conv GEMM: plain K=960 GEMM thanks to contiguous padded halo ----------------
__global__ void __launch_bounds__(256, 1)
k_conv_gemm(const bf16* __restrict__ sent, const bf16* __restrict__ cw,
            const float* __restrict__ cb, bf16* __restrict__ feat) {
  __shared__ bf16 sB[2][64 * BPITCH];
  const int tid  = threadIdx.x;
  const int lane = tid & 31;
  const int w    = tid >> 5;
  const int blockN = blockIdx.x & 3;
  const int blockM = blockIdx.x >> 2;
  const int n0 = blockN << 6;
  const int m0 = (blockM << 8) + (w << 5);   // 256 rows/block, 32 rows/wave
  const int b  = m0 >> 9;
  const int l0 = m0 & 511;
  const int r15  = lane & 15;
  const int aOff = (lane & 16) ? 8 : 0;
  const int bOff = (lane & 16) ? 8 : 0;
  const int sOff = (tid >> 2) * BPITCH + ((tid & 3) << 3);
  const int arow0 = b * LPAD + l0 + r15;     // padded row of l-1 halo start

  const bf16* pA0 = sent + (size_t)arow0 * Dp + aOff;        // K window spans 3 rows
  const bf16* pA1 = pA0 + (size_t)16 * Dp;
  const bf16* pB  = cw + (size_t)(n0 + (tid >> 2)) * KC + ((tid & 3) << 3);

  v8f acc[2][4] = {};
  gemm_core(KC / 32, pA0, pA1, pB, sB[0], sB[1], sOff, lane, bOff, acc);

  const int nc = lane & 15, mAdd = (lane & 16) ? 8 : 0;
#pragma unroll
  for (int t = 0; t < 2; ++t) {
#pragma unroll
    for (int q = 0; q < 4; ++q) {
      int f = n0 + (q << 4) + nc;
      float bias = cb[f];
#pragma unroll
      for (int i = 0; i < 8; ++i) {
        float v = acc[t][q][i] + bias;
        v = v > 0.f ? v : 0.f;               // ReLU
        feat[(size_t)(m0 + (t << 4) + mAdd + i) * F_ + f] = (bf16)v;
      }
    }
  }
}

// ---------------- ROI max pool ----------------
__global__ void k_roi_pool(const bf16* __restrict__ feat, const int* __restrict__ rois,
                           const int* __restrict__ ridx, bf16* __restrict__ pooled) {
  int n = blockIdx.x;
  int f = threadIdx.x;
  int x1 = rois[2 * n], x2 = rois[2 * n + 1];
  int b  = ridx[n];
  const bf16* base = feat + (size_t)b * L_ * F_ + f;
  float m = -3.0e38f;
  for (int l = x1; l < x2; ++l) {
    float v = (float)base[(size_t)l * F_];
    m = fmaxf(m, v);
  }
  pooled[(size_t)n * F_ + f] = (bf16)m;
}

// ---------------- mid GEMM: h[2048][2048] = pooled[2048][256] @ Wc + bias ----------------
__global__ void __launch_bounds__(256, 1)
k_mid_gemm(const bf16* __restrict__ A, const bf16* __restrict__ Bt,
           const float* __restrict__ bias, bf16* __restrict__ out) {
  __shared__ bf16 sB[2][64 * BPITCH];
  const int tid  = threadIdx.x;
  const int lane = tid & 31;
  const int w    = tid >> 5;
  const int blockN = blockIdx.x & 31;        // 32 N-blocks of 64
  const int blockM = blockIdx.x >> 5;        // 8 M-blocks of 256
  const int n0 = blockN << 6;
  const int m0 = (blockM << 8) + (w << 5);
  const int r15  = lane & 15;
  const int aOff = (lane & 16) ? 8 : 0;
  const int bOff = (lane & 16) ? 8 : 0;
  const int sOff = (tid >> 2) * BPITCH + ((tid & 3) << 3);

  const bf16* pA0 = A + (size_t)(m0 + r15) * F_ + aOff;
  const bf16* pA1 = pA0 + (size_t)16 * F_;
  const bf16* pB  = Bt + (size_t)(n0 + (tid >> 2)) * F_ + ((tid & 3) << 3);

  v8f acc[2][4] = {};
  gemm_core(F_ / 32, pA0, pA1, pB, sB[0], sB[1], sOff, lane, bOff, acc);

  const int nc = lane & 15, mAdd = (lane & 16) ? 8 : 0;
#pragma unroll
  for (int t = 0; t < 2; ++t) {
#pragma unroll
    for (int q = 0; q < 4; ++q) {
      int j = n0 + (q << 4) + nc;
      float bv = bias[j];
#pragma unroll
      for (int i = 0; i < 8; ++i)
        out[(size_t)(m0 + (t << 4) + mAdd + i) * FF_ + j] = (bf16)(acc[t][q][i] + bv);
    }
  }
}

// ---------------- head GEMM: out[2048][nValid] = h @ W + bias (K=2048) ----------------
__global__ void __launch_bounds__(256, 1)
k_head_gemm(const bf16* __restrict__ A, const bf16* __restrict__ Bt,
            const float* __restrict__ bias, float* __restrict__ out,
            int nTiles, int nValid) {
  int gtid = blockIdx.x * blockDim.x + threadIdx.x;
  int wid  = gtid >> 5;
  int lane = threadIdx.x & 31;
  int mTile = wid / nTiles;
  int nTile = wid % nTiles;
  int m0 = mTile << 4, n0 = nTile << 4;
  const int aOff = (lane & 16) ? 8 : 0;
  const int bOff = (lane & 16) ? 8 : 0;
  const bf16* Ap = A  + (size_t)(m0 + (lane & 15)) * FF_ + aOff;
  const bf16* Bp = Bt + (size_t)(n0 + (lane & 15)) * FF_ + bOff;

  bf16x8 a0 = *(const bf16x8*)Ap;
  bf16x8 a1 = *(const bf16x8*)(Ap + 16);
  bf16x8 b0 = *(const bf16x8*)Bp;
  bf16x8 b1 = *(const bf16x8*)(Bp + 16);
  v8f acc = {};
  for (int k0 = 0; k0 < FF_; k0 += 32) {
    bf16x16 a = SHUF16(a0, a1);
    bf16x16 b = SHUF16(b0, b1);
    if (k0 + 32 < FF_) {
      a0 = *(const bf16x8*)(Ap + k0 + 32);
      a1 = *(const bf16x8*)(Ap + k0 + 48);
      b0 = *(const bf16x8*)(Bp + k0 + 32);
      b1 = *(const bf16x8*)(Bp + k0 + 48);
    }
    acc = WMMA_BF16(a, b, acc);
  }
  const int nc = lane & 15, mAdd = (lane & 16) ? 8 : 0;
  int col = n0 + nc;
  if (col < nValid) {
    float bv = bias[col];
#pragma unroll
    for (int i = 0; i < 8; ++i)
      out[(size_t)(m0 + mAdd + i) * nValid + col] = acc[i] + bv;
  }
}

// ---------------- host launcher ----------------
extern "C" void kernel_launch(void* const* d_in, const int* in_sizes, int n_in,
                              void* d_out, int out_size, void* d_ws, size_t ws_size,
                              hipStream_t stream) {
  const float* sentence = (const float*)d_in[0];
  const int*   rois     = (const int*)  d_in[1];
  const int*   ridx     = (const int*)  d_in[2];
  const float* conv_w   = (const float*)d_in[3];
  const float* conv_b   = (const float*)d_in[4];
  const float* w1       = (const float*)d_in[5];
  const float* b1       = (const float*)d_in[6];
  const float* wc       = (const float*)d_in[7];
  const float* bc       = (const float*)d_in[8];
  const float* w2       = (const float*)d_in[9];
  const float* b2       = (const float*)d_in[10];
  const float* wb       = (const float*)d_in[11];
  const float* bb       = (const float*)d_in[12];

  char* w = (char*)d_ws;
  // All buffers tail-padded 4KB: gemm_core's pipeline prefetches one K-step past
  // the end of its A/B operands (reads only; values never consumed).
  auto carve = [&](size_t bytes) -> char* {
    char* p = w;
    w += (bytes + 4096 + 255) & ~(size_t)255;
    return p;
  };
  bf16* sent_bf = (bf16*)carve((size_t)B_ * LPAD * Dp * 2);
  bf16* cw_bf   = (bf16*)carve((size_t)F_ * KC * 2);
  bf16* feat_bf = (bf16*)carve((size_t)B_ * L_ * F_ * 2);
  bf16* pooled  = (bf16*)carve((size_t)NROI * F_ * 2);
  bf16* w1cT    = (bf16*)carve((size_t)FF_ * F_ * 2);
  bf16* w2cT    = (bf16*)carve((size_t)FF_ * F_ * 2);
  bf16* wcT     = (bf16*)carve((size_t)32 * FF_ * 2);
  bf16* wbT     = (bf16*)carve((size_t)48 * FF_ * 2);
  bf16* h1      = (bf16*)carve((size_t)NROI * FF_ * 2);
  bf16* h2      = (bf16*)carve((size_t)NROI * FF_ * 2);

  const int T = 256;
  k_prep_sentence<<<(B_ * LPAD * Dp + T - 1) / T, T, 0, stream>>>(sentence, sent_bf);
  k_prep_convw  <<<(F_ * KC + T - 1) / T, T, 0, stream>>>(conv_w, cw_bf);
  k_collapse_w  <<<(FF_ * F_ + T - 1) / T, T, 0, stream>>>(w1, w1cT);
  k_collapse_w  <<<(FF_ * F_ + T - 1) / T, T, 0, stream>>>(w2, w2cT);
  k_transpose_pad<<<(32 * FF_ + T - 1) / T, T, 0, stream>>>(wc, wcT, 32, 19);
  k_transpose_pad<<<(48 * FF_ + T - 1) / T, T, 0, stream>>>(wb, wbT, 48, 38);
  // conv: 128 M-blocks(256 rows) x 4 N-blocks(64 cols)
  k_conv_gemm<<<512, T, 0, stream>>>(sent_bf, cw_bf, conv_b, feat_bf);
  k_roi_pool<<<NROI, F_, 0, stream>>>(feat_bf, rois, ridx, pooled);
  // mid: 8 M-blocks x 32 N-blocks
  k_mid_gemm<<<256, T, 0, stream>>>(pooled, w1cT, b1, h1);
  k_mid_gemm<<<256, T, 0, stream>>>(pooled, w2cT, b2, h2);
  float* cls_out  = (float*)d_out;
  float* bbox_out = cls_out + (size_t)NROI * 19;
  k_head_gemm<<<32, T, 0, stream>>>(h1, wcT, bc, cls_out, 2, 19);
  k_head_gemm<<<48, T, 0, stream>>>(h2, wbT, bb, bbox_out, 3, 38);
}
